// minGRU_56959856280317
// MI455X (gfx1250) — matrix operand order repaired and checked by reading, not compiled, not run on hardware
//
#include <hip/hip_runtime.h>
#include <hip/hip_bf16.h>

// ---------------------------------------------------------------------------
// minGRU on MI455X (gfx1250, wave32):
//   1) convert x / W_hg / W_out fp32 -> bf16
//   2) GEMM1 (bf16 WMMA, f32 accum) + fused epilogue producing linear-space
//      recurrence coefficients a_t = sigmoid(-gate), b_t = sigmoid(gate)*g(hidden)
//   3) chunked parallel scan h_t = a_t*h_{t-1} + b_t (h_{-1} = 1), 3 passes
//   4) GEMM2 (bf16 WMMA) -> f32 output + b_out
//
// GEMM shape: block tile 256(M) x 128(N), 8 waves in a 4(M) x 2(N) grid,
// wave tile 64x64 = 4x4 v_wmma_f32_16x16x32_bf16 accumulators. This gives
// 16 WMMAs per 16 ds_load_b128 fragment fetches per K-step (vs 8:12 for a
// 32x64 tile), keeping the inner loop matrix-issue bound instead of
// LDS-latency bound.
// ---------------------------------------------------------------------------

typedef __attribute__((ext_vector_type(16))) __bf16        v16bf;
typedef __attribute__((ext_vector_type(8)))  float         v8f;
typedef __attribute__((ext_vector_type(4)))  unsigned int  u32x4;

union FragBF { v16bf v; u32x4 q[2]; };

#define BLK_M 256
#define BLK_N 128
#define BLK_K 32
#define LDS_STRIDE 48   // bf16 elements; 96B row pitch -> 16B aligned, bank-skewed

#define NB      4       // batch
#define TT      4096    // time steps
#define DD      512     // model dim
#define DI      2048    // inner dim
#define SCAN_C  64      // chunk length
#define NCHUNK  (TT / SCAN_C)       // 64
#define NCH_TOT (NB * DI)           // 8192 independent channels

// ------------------------------ GEMM (templated) ---------------------------
// EPI == 0 : minGRU front GEMM.  A=[M,512]=x_bf16, Bw=W_hg_bf16 [4096,512].
//            Block covers 64 hidden cols (e) + matching 64 gate cols (e+2048).
//            out0 = a_t, out1 = b_t   ([M, DI] f32 each)
// EPI == 1 : output GEMM. A=[M,2048]=h_bf16, Bw=W_out_bf16 [512,2048].
//            out0 = final f32 output [M, 512]
template <int K, int EPI>
__global__ void __launch_bounds__(256)
gemm_bf16_wmma(const __bf16* __restrict__ A,
               const __bf16* __restrict__ Bw,
               const float*  __restrict__ bias,
               float* __restrict__ out0,
               float* __restrict__ out1)
{
    __shared__ __align__(16) __bf16 As[BLK_M][LDS_STRIDE];
    __shared__ __align__(16) __bf16 Bs[BLK_N][LDS_STRIDE];
    __shared__ float Gs[(EPI == 0) ? BLK_M : 1][68];   // hidden g() exchange

    const int tid   = threadIdx.x;
    const int lane  = tid & 31;
    const int wave  = tid >> 5;        // 0..7
    const int wm    = wave >> 1;       // 0..3 : 64-row strip
    const int wn    = wave & 1;        // 0..1 : 64-col strip
    const int l16   = lane & 15;
    const int lhalf = lane >> 4;       // lane half (WMMA layout)

    const int m0 = blockIdx.y * BLK_M;
    const int nb = blockIdx.x;               // EPI0: e-tile index (64 wide)
    const int n0 = blockIdx.x * BLK_N;       // EPI1: plain col tile

    // A loader: each thread owns one 32-element row (4 x b128)
    const int arow = tid;
    // B loader: two threads per 32-element row (2 x b128)
    const int brow_l = tid >> 1;
    const int bcol_l = (tid & 1) * 16;

    // B-tile global row: EPI0 packs W_hg rows [e .. e+63] and [2048+e .. 2048+e+63]
    const int brow_g = (EPI == 0)
        ? (nb * 64 + (brow_l & 63) + ((brow_l >= 64) ? 2048 : 0))
        : (n0 + brow_l);

    v8f acc[4][4];
#pragma unroll
    for (int mi = 0; mi < 4; ++mi)
#pragma unroll
        for (int ni = 0; ni < 4; ++ni)
#pragma unroll
            for (int r = 0; r < 8; ++r) acc[mi][ni][r] = 0.0f;

    for (int k0 = 0; k0 < K; k0 += BLK_K) {
        const u32x4* gA = (const u32x4*)(A  + (size_t)(m0 + arow) * K + k0);
        const u32x4* gB = (const u32x4*)(Bw + (size_t)brow_g      * K + k0 + bcol_l);
        u32x4 a0 = gA[0], a1 = gA[1], a2 = gA[2], a3 = gA[3];
        u32x4 b0 = gB[0], b1 = gB[1];
        *(u32x4*)&As[arow][0]           = a0;
        *(u32x4*)&As[arow][8]           = a1;
        *(u32x4*)&As[arow][16]          = a2;
        *(u32x4*)&As[arow][24]          = a3;
        *(u32x4*)&Bs[brow_l][bcol_l]     = b0;
        *(u32x4*)&Bs[brow_l][bcol_l + 8] = b1;
        __syncthreads();

        // Build WMMA fragments from LDS per documented CDNA5 layouts.
        FragBF afr[4], bfr[4];
#pragma unroll
        for (int mi = 0; mi < 4; ++mi) {
            const __bf16* p = &As[wm * 64 + mi * 16 + l16][lhalf * 8];
            afr[mi].q[0] = *(const u32x4*)p;        // K = 8h .. 8h+7
            afr[mi].q[1] = *(const u32x4*)(p + 16); // K = 16+8h .. 23+8h
        }
#pragma unroll
        for (int ni = 0; ni < 4; ++ni) {
            const __bf16* p = &Bs[wn * 64 + ni * 16 + l16][lhalf * 8];
            bfr[ni].q[0] = *(const u32x4*)p;        // K = 8h .. 8h+7
            bfr[ni].q[1] = *(const u32x4*)(p + 16); // K = 16+8h .. 23+8h
        }
#pragma unroll
        for (int mi = 0; mi < 4; ++mi)
#pragma unroll
            for (int ni = 0; ni < 4; ++ni)
                acc[mi][ni] = __builtin_amdgcn_wmma_f32_16x16x32_bf16(
                    false, afr[mi].v, false, bfr[ni].v,
                    (short)0, acc[mi][ni], false, false);
        __syncthreads();
    }

    if (EPI == 0) {
        // hidden waves (wn==0): stage g(hidden) through LDS
        if (wn == 0) {
#pragma unroll
            for (int mi = 0; mi < 4; ++mi)
#pragma unroll
                for (int ni = 0; ni < 4; ++ni) {
                    const int el = ni * 16 + l16;
                    const float bv = bias[nb * 64 + el];
#pragma unroll
                    for (int r = 0; r < 8; ++r) {
                        const int ml = wm * 64 + mi * 16 + r + 8 * lhalf;
                        const float v = acc[mi][ni][r] + bv;
                        // g(x) = x+0.5 (x>=0) else sigmoid(x)
                        Gs[ml][el] = (v >= 0.0f) ? (v + 0.5f)
                                                 : (1.0f / (1.0f + __expf(-v)));
                    }
                }
        }
        __syncthreads();
        // gate waves (wn==1): a = sigmoid(-gate), b = sigmoid(gate)*g(hidden)
        if (wn == 1) {
#pragma unroll
            for (int mi = 0; mi < 4; ++mi)
#pragma unroll
                for (int ni = 0; ni < 4; ++ni) {
                    const int el = ni * 16 + l16;
                    const float bv = bias[2048 + nb * 64 + el];
#pragma unroll
                    for (int r = 0; r < 8; ++r) {
                        const int ml = wm * 64 + mi * 16 + r + 8 * lhalf;
                        const float v = acc[mi][ni][r] + bv;
                        const float z = 1.0f / (1.0f + __expf(-v));
                        const size_t o = (size_t)(m0 + ml) * DI + nb * 64 + el;
                        out0[o] = 1.0f - z;          // a_t
                        out1[o] = z * Gs[ml][el];    // b_t
                    }
                }
        }
    } else {
        // plain f32 store + bias
#pragma unroll
        for (int mi = 0; mi < 4; ++mi)
#pragma unroll
            for (int ni = 0; ni < 4; ++ni) {
                const int gn = n0 + wn * 64 + ni * 16 + l16;
                const float bv = bias[gn];
#pragma unroll
                for (int r = 0; r < 8; ++r) {
                    const int gm = m0 + wm * 64 + mi * 16 + r + 8 * lhalf;
                    out0[(size_t)gm * 512 + gn] = acc[mi][ni][r] + bv;
                }
            }
    }
}

// ------------------------------ helpers ------------------------------------
__global__ void cvt_f32_to_bf16(const float* __restrict__ in,
                                __bf16* __restrict__ out, int n)
{
    int i = blockIdx.x * blockDim.x + threadIdx.x;
    if (i < n) out[i] = (__bf16)in[i];
}

// ------------------------------ chunked scan -------------------------------
// Pass A: per (b, chunk, e) compose 64 steps: (A,B) <- (a*A, a*B + b)
__global__ void scan_chunk_reduce(const float* __restrict__ Ab,
                                  const float* __restrict__ Bb,
                                  float* __restrict__ Ach,
                                  float* __restrict__ Bch)
{
    const int tid = blockIdx.x * blockDim.x + threadIdx.x; // NB*NCHUNK*DI
    const int e   = tid & (DI - 1);
    const int rc  = tid >> 11;
    const int c   = rc & (NCHUNK - 1);
    const int b   = rc >> 6;
    size_t base = (size_t)(b * TT + c * SCAN_C) * DI + e;
    float A = 1.0f, Bv = 0.0f;
#pragma unroll 4
    for (int t = 0; t < SCAN_C; ++t) {
        const float a  = Ab[base];
        const float bb = Bb[base];
        A *= a;
        Bv = fmaf(a, Bv, bb);
        base += DI;
    }
    const int o = c * NCH_TOT + b * DI + e;
    Ach[o] = A;
    Bch[o] = Bv;
}

// Pass B: serial carry over 64 chunks per channel; store chunk-entry state.
__global__ void scan_carry(const float* __restrict__ Ach,
                           const float* __restrict__ Bch,
                           float* __restrict__ Hst)
{
    const int tid = blockIdx.x * blockDim.x + threadIdx.x; // NCH_TOT
    float h = 1.0f;                                        // h_{-1} = exp(0)
    for (int c = 0; c < NCHUNK; ++c) {
        const int o = c * NCH_TOT + tid;
        Hst[o] = h;
        h = fmaf(Ach[o], h, Bch[o]);
    }
}

// Pass C: replay each chunk from its entry state, emit h_t as bf16 for GEMM2.
__global__ void scan_apply(const float* __restrict__ Ab,
                           const float* __restrict__ Bb,
                           const float* __restrict__ Hst,
                           __bf16* __restrict__ hout)
{
    const int tid = blockIdx.x * blockDim.x + threadIdx.x;
    const int e   = tid & (DI - 1);
    const int rc  = tid >> 11;
    const int c   = rc & (NCHUNK - 1);
    const int b   = rc >> 6;
    float h = Hst[c * NCH_TOT + b * DI + e];
    size_t base = (size_t)(b * TT + c * SCAN_C) * DI + e;
#pragma unroll 4
    for (int t = 0; t < SCAN_C; ++t) {
        h = fmaf(Ab[base], h, Bb[base]);
        hout[base] = (__bf16)h;
        base += DI;
    }
}

// ------------------------------ launcher -----------------------------------
extern "C" void kernel_launch(void* const* d_in, const int* in_sizes, int n_in,
                              void* d_out, int out_size, void* d_ws, size_t ws_size,
                              hipStream_t stream)
{
    const float* x     = (const float*)d_in[0];   // [4,4096,512]
    const float* W_hg  = (const float*)d_in[1];   // [4096,512]
    const float* b_hg  = (const float*)d_in[2];   // [4096]
    const float* W_out = (const float*)d_in[3];   // [512,2048]
    const float* b_out = (const float*)d_in[4];   // [512]

    const int M = NB * TT;                        // 16384

    char*  ws  = (char*)d_ws;
    size_t off = 0;
    auto alloc = [&](size_t bytes) -> void* {
        void* p = ws + off;
        off += (bytes + 255) & ~(size_t)255;
        return p;
    };

    __bf16* x_bf    = (__bf16*)alloc((size_t)M * DD * 2);        // 16 MB
    __bf16* Whg_bf  = (__bf16*)alloc((size_t)2 * DI * DD * 2);   // 4 MB
    __bf16* Wout_bf = (__bf16*)alloc((size_t)DD * DI * 2);       // 2 MB
    float*  Abuf    = (float*)alloc((size_t)M * DI * 4);         // 128 MB
    float*  Bbuf    = (float*)alloc((size_t)M * DI * 4);         // 128 MB
    __bf16* h_bf    = (__bf16*)alloc((size_t)M * DI * 2);        // 64 MB
    float*  Ach     = (float*)alloc((size_t)NCH_TOT * NCHUNK * 4);
    float*  Bch     = (float*)alloc((size_t)NCH_TOT * NCHUNK * 4);
    float*  Hst     = (float*)alloc((size_t)NCH_TOT * NCHUNK * 4);

    // 1) fp32 -> bf16 operand conversion
    {
        int n;
        n = M * DD;
        cvt_f32_to_bf16<<<(n + 255) / 256, 256, 0, stream>>>(x, x_bf, n);
        n = 2 * DI * DD;
        cvt_f32_to_bf16<<<(n + 255) / 256, 256, 0, stream>>>(W_hg, Whg_bf, n);
        n = DD * DI;
        cvt_f32_to_bf16<<<(n + 255) / 256, 256, 0, stream>>>(W_out, Wout_bf, n);
    }

    // 2) GEMM1 + fused (a_t, b_t) epilogue.  Grid: e-tiles (DI/64) x M-tiles
    gemm_bf16_wmma<DD, 0><<<dim3(DI / 64, M / BLK_M), 256, 0, stream>>>(
        x_bf, Whg_bf, b_hg, Abuf, Bbuf);

    // 3) chunked scan (3 passes)
    scan_chunk_reduce<<<(NB * NCHUNK * DI) / 256, 256, 0, stream>>>(Abuf, Bbuf, Ach, Bch);
    scan_carry<<<NCH_TOT / 256, 256, 0, stream>>>(Ach, Bch, Hst);
    scan_apply<<<(NB * NCHUNK * DI) / 256, 256, 0, stream>>>(Abuf, Bbuf, Hst, h_bf);

    // 4) GEMM2 -> f32 output (+ b_out)
    gemm_bf16_wmma<DI, 1><<<dim3(512 / BLK_N, M / BLK_M), 256, 0, stream>>>(
        h_bf, Wout_bf, b_out, (float*)d_out, nullptr);
}
